// VFLINGModel_11141145166146
// MI455X (gfx1250) — matrix operand order repaired and checked by reading, not compiled, not run on hardware
//
#include <hip/hip_runtime.h>

#define NN 4096
#define EE 131072
#define HH 256
#define CC 10

typedef __attribute__((ext_vector_type(16))) __bf16 bf16x16;
typedef __attribute__((ext_vector_type(8)))  float  floatx8;

struct __attribute__((aligned(16))) U4 { unsigned int a, b, c, d; };
union Frag { U4 q[2]; bf16x16 m; };

__device__ __forceinline__ unsigned short f2bf(float f) {
  return __builtin_bit_cast(unsigned short, (__bf16)f);   // RNE cvt by hardware
}
__device__ __forceinline__ unsigned pk2(float a, float b) {
  return (unsigned)f2bf(a) | ((unsigned)f2bf(b) << 16);
}
__device__ __forceinline__ unsigned lds_off(const void* p) {
  return (unsigned)(unsigned long long)p;   // low 32 bits of generic ptr = LDS offset
}
// GV-mode async copy: 16 bytes per lane, global -> LDS, tracked by ASYNCcnt
__device__ __forceinline__ void async_b128(unsigned lds, const unsigned short* g) {
  asm volatile("global_load_async_to_lds_b128 %0, %1, off"
               :: "v"(lds), "v"((unsigned long long)g) : "memory");
}
template<int NW> __device__ __forceinline__ void wait_async_n() {
  if constexpr (NW == 0)      asm volatile("s_wait_asynccnt 0x0" ::: "memory");
  else if constexpr (NW == 3) asm volatile("s_wait_asynccnt 0x3" ::: "memory");
  else                        asm volatile("s_wait_asynccnt 0x6" ::: "memory");
}

// ---------------------------------------------------------------------------
// bf16 WMMA GEMM:  C[M x Nout](f32) = A[M x K](bf16) * B[Nout x K](bf16)^T
// B pre-transposed so both operands are K-contiguous.
// 8 waves in a 4x2 grid; wave tile = (TM*16) x (TN*16).
// Double-buffered LDS tiles filled with global_load_async_to_lds_b128.
// epi: 0=store, 1=+bias, 2=+bias+relu, 3=sigmoid
// Requires M%BM==0, Nout%BN==0, K%32==0 (true for all call sites).
// ---------------------------------------------------------------------------
template<int BMt, int BNt, int TM, int TN>
__global__ __launch_bounds__(256)
void wmma_gemm(const unsigned short* __restrict__ A, const unsigned short* __restrict__ B,
               float* __restrict__ C, const float* __restrict__ bias,
               int M, int Nout, int K, int epi)
{
  constexpr int BK  = 32;
  constexpr int LA  = BK + 8;                 // padded LDS row stride (ushorts)
  constexpr int LB  = BK + 8;
  constexpr int AT  = BMt * LA;               // ushorts per A buffer
  constexpr int BT  = BNt * LB;
  constexpr int ACH = (BMt * BK / 8) / 256;   // 16B chunks per thread (A)
  constexpr int BCH = (BNt * BK / 8) / 256;   // 16B chunks per thread (B)
  static_assert(BMt == 4 * TM * 16 && BNt == 2 * TN * 16, "wave grid is 4x2");

  __shared__ unsigned short As[2 * AT];
  __shared__ unsigned short Bs[2 * BT];

  const int tid  = threadIdx.x;
  const int mb   = blockIdx.y * BMt;
  const int nb   = blockIdx.x * BNt;

  const int lane = tid & 31;
  const int w    = tid >> 5;
  const int wm   = w >> 1;     // 0..3
  const int wn   = w & 1;      // 0..1
  const int lm   = lane & 15;
  const int lh   = lane >> 4;

  // static per-thread async-copy assignment (BK=32 -> 4 x 16B chunks per row)
  const unsigned short* gA[ACH]; unsigned oA[ACH];
  const unsigned short* gB[BCH]; unsigned oB[BCH];
  #pragma unroll
  for (int j = 0; j < ACH; ++j) {
    int chunk = tid + j * 256;
    int m = chunk >> 2, q = (chunk & 3) << 3;
    gA[j] = A + (size_t)(mb + m) * K + q;
    oA[j] = lds_off(&As[m * LA + q]);
  }
  #pragma unroll
  for (int j = 0; j < BCH; ++j) {
    int chunk = tid + j * 256;
    int m = chunk >> 2, q = (chunk & 3) << 3;
    gB[j] = B + (size_t)(nb + m) * K + q;
    oB[j] = lds_off(&Bs[m * LB + q]);
  }

  auto issue = [&](int buf, int kb) {
    #pragma unroll
    for (int j = 0; j < ACH; ++j) async_b128(oA[j] + buf * (AT * 2), gA[j] + kb);
    #pragma unroll
    for (int j = 0; j < BCH; ++j) async_b128(oB[j] + buf * (BT * 2), gB[j] + kb);
  };

  floatx8 acc[TM][TN] = {};

  const int nk = K / BK;
  issue(0, 0);                                  // prologue: tile 0 in flight
  for (int i = 0; i < nk; ++i) {
    if (i + 1 < nk) {
      issue((i + 1) & 1, (i + 1) * BK);         // prefetch next tile
      wait_async_n<ACH + BCH>();                // retire tile i, keep next in flight
    } else {
      wait_async_n<0>();
    }
    __syncthreads();

    const unsigned short* Ab = As + (i & 1) * AT;
    const unsigned short* Bb = Bs + (i & 1) * BT;

    // fragments per ISA 16-bit A/B layouts (two ds_load_b128 each)
    Frag af[TM], bfr[TN];
    #pragma unroll
    for (int t = 0; t < TM; ++t) {
      const unsigned short* p = &Ab[(wm * TM * 16 + t * 16 + lm) * LA + lh * 8];
      af[t].q[0] = *(const U4*)p;               // K = lh*8 .. +7
      af[t].q[1] = *(const U4*)(p + 16);        // K = 16+lh*8 .. +7
    }
    #pragma unroll
    for (int t = 0; t < TN; ++t) {
      const unsigned short* p = &Bb[(wn * TN * 16 + t * 16 + lm) * LB + lh * 8];
      bfr[t].q[0] = *(const U4*)p;              // K = lh*8 .. +7
      bfr[t].q[1] = *(const U4*)(p + 16);       // K = 16+lh*8 .. +7
    }
    #pragma unroll
    for (int tm = 0; tm < TM; ++tm)
      #pragma unroll
      for (int tn = 0; tn < TN; ++tn)
        acc[tm][tn] = __builtin_amdgcn_wmma_f32_16x16x32_bf16(
            false, af[tm].m, false, bfr[tn].m, (short)0, acc[tm][tn], false, false);
    __syncthreads();
  }

  // epilogue: C/D layout (lane n = lm, VGPR r -> M = r + 8*lh)
  #pragma unroll
  for (int tm = 0; tm < TM; ++tm) {
    #pragma unroll
    for (int tn = 0; tn < TN; ++tn) {
      int col = nb + wn * TN * 16 + tn * 16 + lm;
      float bv = (epi == 1 || epi == 2) ? bias[col] : 0.0f;
      #pragma unroll
      for (int r = 0; r < 8; ++r) {
        int row = mb + wm * TM * 16 + tm * 16 + lh * 8 + r;
        float v = acc[tm][tn][r];
        if (epi == 1 || epi == 2) v += bv;
        if (epi == 2) v = fmaxf(v, 0.0f);
        if (epi == 3) v = 1.0f / (1.0f + __expf(-v));
        C[(size_t)row * Nout + col] = v;
      }
    }
  }
}

// ---------------------------------------------------------------------------
// Conversion kernels (run once per operand, HBM-bound, negligible cost)
// ---------------------------------------------------------------------------
__global__ void cvt_kernel(const float* __restrict__ in, unsigned short* __restrict__ out,
                           int n4) {   // n4 = elements/4
  int i = blockIdx.x * blockDim.x + threadIdx.x;
  if (i >= n4) return;
  const float4 f = *(const float4*)(in + (size_t)i * 4);
  unsigned int* o = (unsigned int*)(out + (size_t)i * 4);
  o[0] = pk2(f.x, f.y);
  o[1] = pk2(f.z, f.w);
}

// weights [K x Nmat] fp32 -> [Nmat x K] bf16
__global__ void cvtT_kernel(const float* __restrict__ in, unsigned short* __restrict__ out,
                            int K, int Nmat) {
  int idx = blockIdx.x * blockDim.x + threadIdx.x;
  if (idx >= K * Nmat) return;
  int n = idx / K, k = idx % K;
  out[idx] = f2bf(in[(size_t)k * Nmat + n]);
}

// ---------------------------------------------------------------------------
// Graph / elementwise kernels
// ---------------------------------------------------------------------------
__global__ void fill_kernel(float* __restrict__ p, float v, int n) {
  int i = blockIdx.x * blockDim.x + threadIdx.x;
  if (i < n) p[i] = v;
}

__global__ void deg_kernel(const int* __restrict__ ei, const float* __restrict__ ew,
                           float* __restrict__ deg, int E) {
  int e = blockIdx.x * blockDim.x + threadIdx.x;
  if (e >= E) return;
  float wv = ew ? ew[e] : 1.0f;
  atomicAdd(&deg[ei[E + e]], wv);
}

__global__ void rsqrt_kernel(float* __restrict__ p, int n) {
  int i = blockIdx.x * blockDim.x + threadIdx.x;
  if (i < n) p[i] = rsqrtf(p[i]);
}

__global__ void norm_kernel(const int* __restrict__ ei, const float* __restrict__ ew,
                            const float* __restrict__ dinv, float* __restrict__ nr, int E) {
  int e = blockIdx.x * blockDim.x + threadIdx.x;
  if (e >= E) return;
  float wv = ew ? ew[e] : 1.0f;
  nr[e] = dinv[ei[e]] * wv * dinv[ei[E + e]];
}

// one thread = one edge x 4 features; agg[dst] += norm * h[src]
__global__ void scatter_kernel(const int* __restrict__ ei, const float* __restrict__ nr,
                               const float* __restrict__ h, float* __restrict__ agg, int E) {
  int idx = blockIdx.x * blockDim.x + threadIdx.x;
  if (idx >= E * (HH / 4)) return;
  int e = idx >> 6;                  // HH/4 == 64
  int c = (idx & 63) << 2;
  int src = ei[e];
  int dst = ei[E + e];
  float wgt = nr[e];
  const float4 hv = *(const float4*)(h + (size_t)src * HH + c);
  float* ap = agg + (size_t)dst * HH + c;
  atomicAdd(ap + 0, wgt * hv.x);
  atomicAdd(ap + 1, wgt * hv.y);
  atomicAdd(ap + 2, wgt * hv.z);
  atomicAdd(ap + 3, wgt * hv.w);
}

// out = agg + dinv^2 * h + b  (optional relu)
__global__ void gcnfin_kernel(const float* __restrict__ agg, const float* __restrict__ h,
                              const float* __restrict__ dinv, const float* __restrict__ b,
                              float* __restrict__ out, int relu) {
  int idx = blockIdx.x * blockDim.x + threadIdx.x;
  if (idx >= NN * HH) return;
  int i = idx >> 8;
  int f = idx & 255;
  float di = dinv[i];
  float v = agg[idx] + di * di * h[idx] + b[f];
  if (relu) v = fmaxf(v, 0.0f);
  out[idx] = v;
}

__global__ void fuse_kernel(const float* __restrict__ att, const float* __restrict__ r0,
                            const float* __restrict__ r1, const float* __restrict__ r2,
                            float* __restrict__ out) {
  int idx = blockIdx.x * blockDim.x + threadIdx.x;
  if (idx >= NN * HH) return;
  float a0 = att[0], a1 = att[1], a2 = att[2];
  float m  = fmaxf(a0, fmaxf(a1, a2));
  float e0 = __expf(a0 - m), e1 = __expf(a1 - m), e2 = __expf(a2 - m);
  float inv = 1.0f / (e0 + e1 + e2);
  out[idx] = (e0 * r0[idx] + e1 * r1[idx] + e2 * r2[idx]) * inv;
}

// logits = c1[N x 128] @ W[128 x 10] + b  (tiny: plain VALU kernel)
__global__ void cls2_kernel(const float* __restrict__ c1, const float* __restrict__ W,
                            const float* __restrict__ b, float* __restrict__ out) {
  int idx = blockIdx.x * blockDim.x + threadIdx.x;
  if (idx >= NN * CC) return;
  int i = idx / CC, c = idx % CC;
  float acc = b[c];
  const float* row = c1 + (size_t)i * (HH / 2);
  #pragma unroll 8
  for (int k = 0; k < HH / 2; ++k) acc += row[k] * W[k * CC + c];
  out[idx] = acc;
}

// ---------------------------------------------------------------------------
extern "C" void kernel_launch(void* const* d_in, const int* in_sizes, int n_in,
                              void* d_out, int out_size, void* d_ws, size_t ws_size,
                              hipStream_t stream)
{
  (void)in_sizes; (void)n_in; (void)out_size; (void)ws_size;

  const float* xs[3]  = {(const float*)d_in[0], (const float*)d_in[4], (const float*)d_in[8]};
  const int*   eis[3] = {(const int*)d_in[1],   (const int*)d_in[5],   (const int*)d_in[9]};
  const float* ews[3] = {(const float*)d_in[2], (const float*)d_in[6], (const float*)d_in[10]};
  const int    Kd[3]  = {512, 256, 128};

  const float* att    = (const float*)d_in[12];
  const float* encW1  = (const float*)d_in[13];
  const float* encb1  = (const float*)d_in[14];
  const float* encW2  = (const float*)d_in[15];
  const float* encb2  = (const float*)d_in[16];
  const float* decW[3]= {(const float*)d_in[17], (const float*)d_in[18], (const float*)d_in[19]};
  const float* clsW1  = (const float*)d_in[20];
  const float* clsb1  = (const float*)d_in[21];
  const float* clsW2  = (const float*)d_in[22];
  const float* clsb2  = (const float*)d_in[23];
  const float* vW1[3] = {(const float*)d_in[24], (const float*)d_in[28], (const float*)d_in[32]};
  const float* vb1[3] = {(const float*)d_in[25], (const float*)d_in[29], (const float*)d_in[33]};
  const float* vW2[3] = {(const float*)d_in[26], (const float*)d_in[30], (const float*)d_in[34]};
  const float* vb2[3] = {(const float*)d_in[27], (const float*)d_in[31], (const float*)d_in[35]};

  float* wp   = (float*)d_ws;
  float* dinv = wp;  wp += 4 * NN;                 // per-view dinv + refiner dinv
  float* nrm  = wp;  wp += 4 * (size_t)EE;         // per-edge norms
  float* hbuf = wp;  wp += (size_t)NN * HH;        // h = x @ W
  float* agg  = wp;  wp += (size_t)NN * HH;        // scatter accumulator
  float* tbuf = wp;  wp += (size_t)NN * HH;        // layer intermediates / fused
  float* zbuf = wp;  wp += (size_t)NN * HH;        // refined embedding z (f32)
  float* Pbuf = wp;  wp += (size_t)NN * HH;        // z @ dec_W (f32)
  float* c1   = wp;  wp += (size_t)NN * (HH / 2);  // classifier hidden
  unsigned short* bfA = (unsigned short*)wp;  wp += (size_t)NN * 512 / 2;  // A operand bf16
  unsigned short* zbf = (unsigned short*)wp;  wp += (size_t)NN * HH / 2;   // z bf16 (A & B role)
  unsigned short* bfB = (unsigned short*)wp;  wp += (512 * HH) / 2;        // weight bf16 [N][K]

  float* out        = (float*)d_out;
  float* out_logits = out;                              // N*C
  float* out_recon  = out + (size_t)NN * CC;            // 3 * N*N
  float* out_rep    = out_recon + 3 * (size_t)NN * NN;  // 3 * N*H

  const int T = 256;
  auto blocks = [](long long n) { return (int)((n + 255) / 256); };

  // --- degrees (with self-loop), dinv = rsqrt, per-edge sym norms ---
  fill_kernel<<<blocks(4 * NN), T, 0, stream>>>(dinv, 1.0f, 4 * NN);
  for (int v = 0; v < 3; ++v)
    deg_kernel<<<blocks(EE), T, 0, stream>>>(eis[v], ews[v], dinv + v * NN, EE);
  deg_kernel<<<blocks(EE), T, 0, stream>>>(eis[0], nullptr, dinv + 3 * NN, EE);
  rsqrt_kernel<<<blocks(4 * NN), T, 0, stream>>>(dinv, 4 * NN);
  for (int v = 0; v < 3; ++v)
    norm_kernel<<<blocks(EE), T, 0, stream>>>(eis[v], ews[v], dinv + v * NN,
                                              nrm + (size_t)v * EE, EE);
  norm_kernel<<<blocks(EE), T, 0, stream>>>(eis[0], nullptr, dinv + 3 * NN,
                                            nrm + 3 * (size_t)EE, EE);

  // C = A(f32,MxK) @ W(f32,KxNout) via bf16 staging buffers (small-tile variant)
  auto gemm_f32 = [&](const float* Af, const float* Wf, float* Cf, const float* bias,
                      int M, int Nout, int K, int epi) {
    cvt_kernel<<<blocks((long long)M * K / 4), T, 0, stream>>>(Af, bfA, M * K / 4);
    cvtT_kernel<<<blocks((long long)K * Nout), T, 0, stream>>>(Wf, bfB, K, Nout);
    wmma_gemm<128, 64, 2, 2><<<dim3(Nout / 64, M / 128), T, 0, stream>>>(
        bfA, bfB, Cf, bias, M, Nout, K, epi);
  };

  auto gcn_layer = [&](const float* xin, int K, const float* W, const float* bvec,
                       const int* ei, const float* nr, const float* dv,
                       float* dst, int relu) {
    gemm_f32(xin, W, hbuf, nullptr, NN, HH, K, 0);
    fill_kernel<<<blocks((long long)NN * HH), T, 0, stream>>>(agg, 0.0f, NN * HH);
    scatter_kernel<<<blocks((long long)EE * (HH / 4)), T, 0, stream>>>(ei, nr, hbuf, agg, EE);
    gcnfin_kernel<<<blocks((long long)NN * HH), T, 0, stream>>>(agg, hbuf, dv, bvec, dst, relu);
  };

  // --- per-view 2-layer GCN encoders (view_reps written straight to d_out) ---
  for (int v = 0; v < 3; ++v) {
    gcn_layer(xs[v], Kd[v], vW1[v], vb1[v], eis[v], nrm + (size_t)v * EE,
              dinv + v * NN, tbuf, 1);
    gcn_layer(tbuf, HH, vW2[v], vb2[v], eis[v], nrm + (size_t)v * EE,
              dinv + v * NN, out_rep + (size_t)v * NN * HH, 0);
  }

  // --- softmax-attention fusion ---
  fuse_kernel<<<blocks((long long)NN * HH), T, 0, stream>>>(
      att, out_rep, out_rep + (size_t)NN * HH, out_rep + 2 * (size_t)NN * HH, tbuf);

  // --- refiner GCN on graph 0 with unit edge weights ---
  gcn_layer(tbuf, HH, encW1, encb1, eis[0], nrm + 3 * (size_t)EE, dinv + 3 * NN, tbuf, 1);
  gcn_layer(tbuf, HH, encW2, encb2, eis[0], nrm + 3 * (size_t)EE, dinv + 3 * NN, zbuf, 0);

  // --- dense decoders: sigmoid((z @ Wv) @ z^T), 4096x4096x256 each ---
  cvt_kernel<<<blocks((long long)NN * HH / 4), T, 0, stream>>>(zbuf, zbf, NN * HH / 4);
  for (int v = 0; v < 3; ++v) {
    cvtT_kernel<<<blocks((long long)HH * HH), T, 0, stream>>>(decW[v], bfB, HH, HH);
    wmma_gemm<128, 64, 2, 2><<<dim3(HH / 64, NN / 128), T, 0, stream>>>(
        zbf, bfB, Pbuf, nullptr, NN, HH, HH, 0);
    cvt_kernel<<<blocks((long long)NN * HH / 4), T, 0, stream>>>(Pbuf, bfA, NN * HH / 4);
    // big-tile variant: 64x64 per wave, 16 WMMA per 16 LDS fragment loads
    wmma_gemm<256, 128, 4, 4><<<dim3(NN / 128, NN / 256), T, 0, stream>>>(
        bfA, zbf, out_recon + (size_t)v * NN * NN, nullptr, NN, NN, HH, 3);
  }

  // --- classifier ---
  cvtT_kernel<<<blocks((long long)HH * (HH / 2)), T, 0, stream>>>(clsW1, bfB, HH, HH / 2);
  wmma_gemm<128, 64, 2, 2><<<dim3((HH / 2) / 64, NN / 128), T, 0, stream>>>(
      zbf, bfB, c1, clsb1, NN, HH / 2, HH, 2);
  cls2_kernel<<<blocks((long long)NN * CC), T, 0, stream>>>(c1, clsW2, clsb2, out_logits);
}
